// AET_34737695490187
// MI455X (gfx1250) — matrix operand order repaired and checked by reading, not compiled, not run on hardware
//
#include <hip/hip_runtime.h>
#include <math.h>

#define BB   64
#define LL   512
#define CC   768
#define PP   196
#define PPAD 208   // 13 * 16, padded row count
#define NT   13

typedef __attribute__((ext_vector_type(8)))  __bf16 v8bf;
typedef __attribute__((ext_vector_type(16))) __bf16 v16bf;
typedef __attribute__((ext_vector_type(8)))  float  v8f;

// ---------------------------------------------------------------- labels
__global__ void AET_labels_k(const int* __restrict__ bbox,
                             const int* __restrict__ att,
                             int* __restrict__ lab) {
    int i = blockIdx.x * blockDim.x + threadIdx.x;
    if (i >= BB * LL) return;
    int x0 = bbox[4 * i + 0] / 72;
    int y0 = bbox[4 * i + 1] / 72;
    int x1 = bbox[4 * i + 2] / 72;
    int y1 = bbox[4 * i + 3] / 72;
    int l = ((x0 == x1) & (y0 == y1) & (att[i] != 0)) ? (y0 * 14 + x0) : -100;
    lab[i] = l;
}

// ---------------------------------------- ipe fp32 -> bf16, rows padded to 208
__global__ void AET_cvt_k(const float* __restrict__ ipe, __bf16* __restrict__ out) {
    int i = blockIdx.x * blockDim.x + threadIdx.x;
    const int total = BB * PPAD * CC;
    if (i >= total) return;
    int c   = i % CC;
    int br  = i / CC;
    int row = br % PPAD;
    int b   = br / PPAD;
    float v = 0.0f;
    if (row < PP) v = ipe[((size_t)b * PP + row) * CC + c];
    out[i] = (__bf16)v;
}

// -------------------------- segmented mean -> tpe (bf16, rows padded to 208)
// One block per (b, p). Labels staged in LDS; branch is block-uniform, so
// each matched embedding row is read exactly once per owning patch.
__global__ void AET_tpe_k(const float* __restrict__ te,
                          const int* __restrict__ lab,
                          __bf16* __restrict__ tpe) {
    int bp = blockIdx.x;
    int b = bp / PPAD, p = bp % PPAD;
    int t = threadIdx.x;
    __bf16* o = tpe + ((size_t)b * PPAD + p) * CC;
    if (p >= PP) {                       // zero-fill padding rows
        o[t] = (__bf16)0.0f;
        o[t + 256] = (__bf16)0.0f;
        o[t + 512] = (__bf16)0.0f;
        return;
    }
    __shared__ int slab[LL];
    for (int l = t; l < LL; l += blockDim.x) slab[l] = lab[b * LL + l];
    __syncthreads();
    const float* base = te + (size_t)b * LL * CC;
    float a0 = 0.f, a1 = 0.f, a2 = 0.f;
    int cnt = 0;
    for (int l = 0; l < LL; ++l) {
        if (slab[l] == p) {
            const float* r = base + (size_t)l * CC;
            ++cnt;
            a0 += r[t];
            a1 += r[t + 256];
            a2 += r[t + 512];
        }
    }
    float inv = 1.0f / fmaxf((float)cnt, 1.0f);
    o[t]       = (__bf16)(a0 * inv);
    o[t + 256] = (__bf16)(a1 * inv);
    o[t + 512] = (__bf16)(a2 * inv);
}

// -------------------------------------------- batched GEMM via bf16 WMMA
// M[b] (208x208, valid 196x196) = A[b] (208x768 bf16) . B[b]^T (bf16).
// One wave32 per 16x16 tile; K-loop: 4 x b128 loads + 1 x wmma, branch-free.
__global__ __launch_bounds__(32) void AET_gemm_k(const __bf16* __restrict__ A,
                                                 const __bf16* __restrict__ Bm,
                                                 float* __restrict__ Mout) {
    int b  = blockIdx.z;
    int m0 = blockIdx.y * 16;
    int n0 = blockIdx.x * 16;
    int lane = threadIdx.x & 31;
    int half = lane >> 4;
    int idx  = lane & 15;

    // A 16-bit layout (ISA 7.12.2): half h holds row (lane&15), K = h*8 + {0..7,16..23}
    const __bf16* arow = A  + ((size_t)b * PPAD + m0 + idx) * CC + half * 8;
    // B 32x16 layout: half h holds col (lane&15), K = h*8 + {0..7,16..23}
    const __bf16* brow = Bm + ((size_t)b * PPAD + n0 + idx) * CC + half * 8;

    v8f c = {};
    for (int k0 = 0; k0 < CC; k0 += 32) {
        v8bf a_lo = *(const v8bf*)(arow + k0);
        v8bf a_hi = *(const v8bf*)(arow + k0 + 16);
        v8bf b_lo = *(const v8bf*)(brow + k0);
        v8bf b_hi = *(const v8bf*)(brow + k0 + 16);
        v16bf a  = __builtin_shufflevector(a_lo, a_hi,
                     0, 1, 2, 3, 4, 5, 6, 7, 8, 9, 10, 11, 12, 13, 14, 15);
        v16bf bf = __builtin_shufflevector(b_lo, b_hi,
                     0, 1, 2, 3, 4, 5, 6, 7, 8, 9, 10, 11, 12, 13, 14, 15);
        c = __builtin_amdgcn_wmma_f32_16x16x32_bf16(
                /*neg_a=*/false, a, /*neg_b=*/false, bf,
                /*c_mod=*/(short)0, c, /*reuse_a=*/false, /*reuse_b=*/false);
    }

#pragma unroll
    for (int r = 0; r < 8; ++r) {
        int m = m0 + r + half * 8;    // C/D layout: VGPR r -> row r (+8 for hi half)
        Mout[((size_t)b * PPAD + m) * PPAD + n0 + idx] = c[r];
    }
}

// ------------------------------------- log-sum-exp over rows and columns
__global__ void AET_lse_k(const float* __restrict__ M, float* __restrict__ lse) {
    int bp = blockIdx.x;
    int b = bp / PP, p = bp % PP;
    int colmode = blockIdx.y;          // 0 = row LSE, 1 = column LSE
    int t = threadIdx.x;
    const float* base = M + (size_t)b * PPAD * PPAD;

    float v = -__builtin_inff();
    if (t < PP) v = colmode ? base[(size_t)t * PPAD + p] : base[(size_t)p * PPAD + t];

    __shared__ float sh[256];
    sh[t] = v; __syncthreads();
    for (int s = 128; s > 0; s >>= 1) {
        if (t < s) sh[t] = fmaxf(sh[t], sh[t + s]);
        __syncthreads();
    }
    float mx = sh[0]; __syncthreads();
    float e = (t < PP) ? expf(v - mx) : 0.0f;
    sh[t] = e; __syncthreads();
    for (int s = 128; s > 0; s >>= 1) {
        if (t < s) sh[t] += sh[t + s];
        __syncthreads();
    }
    if (t == 0) lse[(size_t)colmode * (BB * PP) + bp] = mx + logf(sh[0]);
}

// ----------------------------------------------------- final scalar loss
__global__ void AET_loss_k(const float* __restrict__ M,
                           const float* __restrict__ lse,
                           float* __restrict__ out) {
    __shared__ float sh[256];
    int t = threadIdx.x;
    float acc = 0.0f;
    for (int i = t; i < BB * PP; i += 256) {
        int b = i / PP, p = i % PP;
        float d = M[((size_t)b * PPAD + p) * PPAD + p];
        acc += 0.5f * (lse[i] + lse[BB * PP + i]) - d;
    }
    sh[t] = acc; __syncthreads();
    for (int s = 128; s > 0; s >>= 1) {
        if (t < s) sh[t] += sh[t + s];
        __syncthreads();
    }
    if (t == 0) out[0] = sh[0] / (float)(BB * PP);
}

extern "C" void kernel_launch(void* const* d_in, const int* in_sizes, int n_in,
                              void* d_out, int out_size, void* d_ws, size_t ws_size,
                              hipStream_t stream) {
    const float* te  = (const float*)d_in[0];   // text_embeds        [B,L,C] f32
    const float* ipe = (const float*)d_in[1];   // image patch embeds [B,P,C] f32
    const int*   bbx = (const int*)d_in[2];     // bbox               [B,L,4] int
    const int*   att = (const int*)d_in[3];     // attention_mask     [B,L]   int
    float* out = (float*)d_out;

    // workspace carve-up (all offsets 256B aligned)
    char* ws = (char*)d_ws;
    size_t off = 0;
    int*    lab   = (int*)(ws + off);    off += (size_t)BB * LL * sizeof(int);
    off = (off + 255) & ~(size_t)255;
    __bf16* ipeb  = (__bf16*)(ws + off); off += (size_t)BB * PPAD * CC * sizeof(__bf16); // 20.4 MB
    off = (off + 255) & ~(size_t)255;
    __bf16* tpeb  = (__bf16*)(ws + off); off += (size_t)BB * PPAD * CC * sizeof(__bf16); // 20.4 MB
    off = (off + 255) & ~(size_t)255;
    float*  Mw    = (float*)(ws + off);  off += (size_t)BB * PPAD * PPAD * sizeof(float); // 11.1 MB
    off = (off + 255) & ~(size_t)255;
    float*  lse   = (float*)(ws + off);                                                   // 100 KB

    AET_labels_k<<<(BB * LL + 255) / 256, 256, 0, stream>>>(bbx, att, lab);
    AET_cvt_k<<<(BB * PPAD * CC + 255) / 256, 256, 0, stream>>>(ipe, ipeb);
    AET_tpe_k<<<BB * PPAD, 256, 0, stream>>>(te, lab, tpeb);
    AET_gemm_k<<<dim3(NT, NT, BB), 32, 0, stream>>>(ipeb, tpeb, Mw);
    AET_lse_k<<<dim3(BB * PP, 2), 256, 0, stream>>>(Mw, lse);
    AET_loss_k<<<1, 256, 0, stream>>>(Mw, lse, out);
}
